// MGAT_90881507983366
// MI455X (gfx1250) — matrix operand order repaired and checked by reading, not compiled, not run on hardware
//
#include <hip/hip_runtime.h>
#include <hip/hip_bf16.h>
#include <math.h>
#include <stdint.h>

typedef __attribute__((ext_vector_type(16))) _Float16 v16h;
typedef __attribute__((ext_vector_type(8)))  _Float16 v8h;
typedef __attribute__((ext_vector_type(4)))  _Float16 v4h;
typedef __attribute__((ext_vector_type(8)))  float    v8f;

#define NEG_SLOPE 0.2f
#define DIMC 128   // all per-node feature widths in this model are 128 (= HEADS*64)

// ---------------------------------------------------------------------------
// Ordered-int float atomic max (deterministic result regardless of order)
// ---------------------------------------------------------------------------
__device__ __forceinline__ void atomicMaxF(float* addr, float v) {
    if (v >= 0.0f) {
        atomicMax((int*)addr, __float_as_int(v));
    } else {
        atomicMin((unsigned int*)addr, (unsigned int)__float_as_int(v));
    }
}

// ---------------------------------------------------------------------------
// f32 -> f16 streaming conversion, 4 elements/thread (b128 in, b64 out)
// ---------------------------------------------------------------------------
__global__ void gatv2_cvt_f16(const float* __restrict__ src, _Float16* __restrict__ dst,
                              long long n4)
{
    const long long t = (long long)blockIdx.x * blockDim.x + threadIdx.x;
    if (t >= n4) return;
    const float4 v = ((const float4*)src)[t];
    v4h h = { (_Float16)v.x, (_Float16)v.y, (_Float16)v.z, (_Float16)v.w };
    ((v4h*)dst)[t] = h;
}

// ---------------------------------------------------------------------------
// Transposed f16 copies of both weight matrices: WT[col][k] = (f16)W[k][col].
// One thread per output element; 128x128 each.
// ---------------------------------------------------------------------------
__global__ void gatv2_cvt_wT(const float* __restrict__ Wl, const float* __restrict__ Wr,
                             _Float16* __restrict__ WTl, _Float16* __restrict__ WTr)
{
    const int t   = blockIdx.x * blockDim.x + threadIdx.x;   // t = col*128 + k
    const int col = t >> 7;
    const int k   = t & 127;
    WTl[t] = (_Float16)Wl[k * DIMC + col];
    WTr[t] = (_Float16)Wr[k * DIMC + col];
}

// ---------------------------------------------------------------------------
// GEMM: out = Xh[N,128] @ WT^T + b, for both (WTl,bl)->xl and (WTr,br)->xr.
// One wave computes one 16x16 f32 tile: 16x global_load_b128 + 4x
// v_wmma_f32_16x16x32_f16, no conversions in the loop. Full tiles take a
// uniform unguarded store path (no exec-mask churn in the epilogue).
// blockDim.x = 256 (8 waves). grid = (ceil(N/128), 8, 2); blockIdx.z picks l/r.
// ---------------------------------------------------------------------------
__global__ void gatv2_gemm(const _Float16* __restrict__ Xh,
                           const _Float16* __restrict__ WTl, const float* __restrict__ bl,
                           const _Float16* __restrict__ WTr, const float* __restrict__ br,
                           float* __restrict__ xl, float* __restrict__ xr, int N)
{
    const int wave = threadIdx.x >> 5;
    const int lane = threadIdx.x & 31;
    const int m0   = (blockIdx.x * (blockDim.x >> 5) + wave) * 16;
    if (m0 >= N) return;                       // wave-uniform: EXEC all-ones for WMMA
    const bool fullTile = (m0 + 16 <= N);      // wave-uniform

    const int n0   = blockIdx.y * 16;
    const _Float16* __restrict__ WT   = blockIdx.z ? WTr : WTl;
    const float*    __restrict__ bias = blockIdx.z ? br : bl;
    float*          __restrict__ out  = blockIdx.z ? xr : xl;

    const int row  = lane & 15;                // A row (M) held by this lane
    const int half = lane >> 4;                // K-half selector
    const int col  = n0 + (lane & 15);         // B/C column (N) held by this lane

    int arow = m0 + row;
    if (arow >= N) arow = N - 1;               // clamp (tail tiles only; store guarded)
    const _Float16* __restrict__ Arow = Xh + (size_t)arow * DIMC;
    const _Float16* __restrict__ Bcol = WT + (size_t)col * DIMC;

    v8f acc = {};
    #pragma unroll
    for (int kb = 0; kb < DIMC; kb += 32) {
        // 16-bit A 16x32 layout: elems 0-7 -> K = kb+8*half+{0..7},
        //                        elems 8-15 -> K = kb+16+8*half+{0..7}
        union { v16h v; v8h h[2]; } a, b;
        a.h[0] = *(const v8h*)(Arow + kb + 8 * half);
        a.h[1] = *(const v8h*)(Arow + kb + 16 + 8 * half);
        // 16-bit B 32x16 layout: same K order as A (elems 0-7 -> kb+8*half+{0..7}, 8-15 -> kb+16+8*half+{0..7})
        b.h[0] = *(const v8h*)(Bcol + kb + 8 * half);
        b.h[1] = *(const v8h*)(Bcol + kb + 16 + 8 * half);
        acc = __builtin_amdgcn_wmma_f32_16x16x32_f16(
                /*neg_a=*/false, a.v, /*neg_b=*/false, b.v,
                /*c_mod=*/(short)0, acc, /*reuse_a=*/false, /*reuse_b=*/false);
    }

    // C/D layout: VGPR i -> M = i (lanes 0-15) or i+8 (lanes 16-31)
    const float bv = bias[col];
    float* outCol = out + (size_t)(m0 + 8 * half) * DIMC + col;
    if (fullTile) {
        #pragma unroll
        for (int i = 0; i < 8; ++i)
            outCol[(size_t)i * DIMC] = acc[i] + bv;
    } else {
        #pragma unroll
        for (int i = 0; i < 8; ++i)
            if (m0 + i + 8 * half < N) outCol[(size_t)i * DIMC] = acc[i] + bv;
    }
}

// ---------------------------------------------------------------------------
// Re-initialize per-layer accumulators (harness does not re-zero workspace).
// ---------------------------------------------------------------------------
__global__ void gatv2_init(float* __restrict__ agg, float* __restrict__ m,
                           float* __restrict__ denom, int Nn)
{
    const long long i = (long long)blockIdx.x * blockDim.x + threadIdx.x;
    if (i < (long long)Nn * DIMC) agg[i] = 0.0f;
    if (i < 2LL * Nn) { m[i] = -__builtin_inff(); denom[i] = 0.0f; }
}

// ---------------------------------------------------------------------------
// Pass 1: per-edge attention logits + running segment max over dst.
// One wave per edge; lane l covers channels 4l..4l+3 (lanes 0-15: head 0,
// lanes 16-31: head 1). Half-wave butterfly reduce.
// ---------------------------------------------------------------------------
__global__ void gatv2_logits(const float* __restrict__ xl, const float* __restrict__ xr,
                             const float* __restrict__ att, const int* __restrict__ ei,
                             int E, int Nn, float* __restrict__ logit, float* __restrict__ m)
{
    const int wave = threadIdx.x >> 5;
    const int lane = threadIdx.x & 31;
    const long long e = (long long)blockIdx.x * (blockDim.x >> 5) + wave;
    const long long Etot = (long long)E + Nn;
    if (e >= Etot) return;

    int src, dst;
    if (e < E) { src = ei[e]; dst = ei[(long long)E + e]; }
    else       { src = dst = (int)(e - E); }

    const float4 xlv = *(const float4*)(xl + (size_t)src * DIMC + lane * 4);
    const float4 xrv = *(const float4*)(xr + (size_t)dst * DIMC + lane * 4);
    const float4 av  = *(const float4*)(att + lane * 4);    // att flat [2,64] == 128

    float p = 0.0f;
    {
        float s;
        s = xlv.x + xrv.x; p += av.x * (s > 0.0f ? s : NEG_SLOPE * s);
        s = xlv.y + xrv.y; p += av.y * (s > 0.0f ? s : NEG_SLOPE * s);
        s = xlv.z + xrv.z; p += av.z * (s > 0.0f ? s : NEG_SLOPE * s);
        s = xlv.w + xrv.w; p += av.w * (s > 0.0f ? s : NEG_SLOPE * s);
    }
    // reduce within each half-wave (head) — masks < 16 keep heads separate
    p += __shfl_xor(p, 1, 32);
    p += __shfl_xor(p, 2, 32);
    p += __shfl_xor(p, 4, 32);
    p += __shfl_xor(p, 8, 32);

    if ((lane & 15) == 0) {
        const int hh = lane >> 4;
        logit[e * 2 + hh] = p;
        atomicMaxF(&m[(size_t)dst * 2 + hh], p);
    }
}

// ---------------------------------------------------------------------------
// Pass 2: ex = exp(logit - m[dst]) (in-place) and denominator segment sum.
// One thread per (edge, head).
// ---------------------------------------------------------------------------
__global__ void gatv2_exp(float* __restrict__ logit, const float* __restrict__ m,
                          const int* __restrict__ ei, int E, int Nn,
                          float* __restrict__ denom)
{
    const long long t = (long long)blockIdx.x * blockDim.x + threadIdx.x;
    const long long Etot2 = 2LL * ((long long)E + Nn);
    if (t >= Etot2) return;

    const long long e = t >> 1;
    const int hh = (int)(t & 1);
    const int dst = (e < E) ? ei[(long long)E + e] : (int)(e - E);

    const float ex = expf(logit[t] - m[(size_t)dst * 2 + hh]);
    logit[t] = ex;
    atomicAdd(&denom[(size_t)dst * 2 + hh], ex);
}

// ---------------------------------------------------------------------------
// Pass 3: alpha-weighted scatter of xl[src] into agg[dst]. One wave per edge.
// ---------------------------------------------------------------------------
__global__ void gatv2_scatter(const float* __restrict__ xl, const float* __restrict__ ex,
                              const float* __restrict__ denom, const int* __restrict__ ei,
                              int E, int Nn, float* __restrict__ agg)
{
    const int wave = threadIdx.x >> 5;
    const int lane = threadIdx.x & 31;
    const long long e = (long long)blockIdx.x * (blockDim.x >> 5) + wave;
    const long long Etot = (long long)E + Nn;
    if (e >= Etot) return;

    int src, dst;
    if (e < E) { src = ei[e]; dst = ei[(long long)E + e]; }
    else       { src = dst = (int)(e - E); }

    const int hh = lane >> 4;
    const float alpha = ex[e * 2 + hh] / (denom[(size_t)dst * 2 + hh] + 1e-16f);

    const float4 v = *(const float4*)(xl + (size_t)src * DIMC + lane * 4);
    float* base = agg + (size_t)dst * DIMC + lane * 4;
    atomicAdd(base + 0, v.x * alpha);
    atomicAdd(base + 1, v.y * alpha);
    atomicAdd(base + 2, v.z * alpha);
    atomicAdd(base + 3, v.w * alpha);
}

// ---------------------------------------------------------------------------
// Finalize: out = ELU(agg + bias)
// ---------------------------------------------------------------------------
__global__ void gatv2_final(const float* __restrict__ agg, const float* __restrict__ bias,
                            float* __restrict__ out, long long n)
{
    const long long i = (long long)blockIdx.x * blockDim.x + threadIdx.x;
    if (i >= n) return;
    const float x = agg[i] + bias[i & (DIMC - 1)];
    out[i] = x > 0.0f ? x : (expf(x) - 1.0f);
}

// ---------------------------------------------------------------------------
// Host launcher
// ---------------------------------------------------------------------------
extern "C" void kernel_launch(void* const* d_in, const int* in_sizes, int n_in,
                              void* d_out, int out_size, void* d_ws, size_t ws_size,
                              hipStream_t stream)
{
    const float* X  = (const float*)d_in[0];
    const int*   ei = (const int*)  d_in[1];
    const int N = in_sizes[0] / DIMC;     // 50000
    const int E = in_sizes[2];            // edge_weight length = 800000
    const long long Etot = (long long)E + N;
    const long long n128 = (long long)N * DIMC;

    // workspace layout (floats)
    float* ws    = (float*)d_ws;
    float* xl    = ws;
    float* xr    = ws + n128;
    float* agg   = ws + 2 * n128;
    float* logit = ws + 3 * n128;
    float* m     = logit + 2 * Etot;
    float* denom = m + 2LL * N;
    // f16 scratch: Xh aliases agg (only live before agg is zeroed); WT after denom
    _Float16* Xh  = (_Float16*)agg;                  // n128 halves <= n128 floats
    _Float16* WTl = (_Float16*)(((uintptr_t)(denom + 2LL * N) + 63) & ~(uintptr_t)63);
    _Float16* WTr = WTl + DIMC * DIMC;

    const int blk = 256;
    const dim3 gemmGrid((unsigned)((N + 127) / 128), 8, 2);            // 8 waves/block, 16 rows/wave
    const unsigned edgeWaveBlocks = (unsigned)((Etot + 7) / 8);         // 1 wave per edge
    const unsigned ehBlocks       = (unsigned)((2 * Etot + blk - 1) / blk);
    const unsigned initBlocks     = (unsigned)((n128 + blk - 1) / blk);
    const unsigned cvtBlocks      = (unsigned)((n128 / 4 + blk - 1) / blk);
    const unsigned wtBlocks       = (unsigned)((DIMC * DIMC + blk - 1) / blk);

    for (int layer = 0; layer < 2; ++layer) {
        const float* h    = (layer == 0) ? X : (const float*)d_out;     // layer 2 reads h1
        const int    base = (layer == 0) ? 4 : 10;
        const float* Wl   = (const float*)d_in[base + 0];
        const float* bl   = (const float*)d_in[base + 1];
        const float* Wr   = (const float*)d_in[base + 2];
        const float* br   = (const float*)d_in[base + 3];
        const float* att  = (const float*)d_in[base + 4];
        const float* bias = (const float*)d_in[base + 5];
        float* outp = (float*)d_out + (long long)layer * n128;

        gatv2_cvt_f16<<<cvtBlocks,      blk, 0, stream>>>(h, Xh, n128 / 4);
        gatv2_cvt_wT <<<wtBlocks,       blk, 0, stream>>>(Wl, Wr, WTl, WTr);
        gatv2_gemm   <<<gemmGrid,       blk, 0, stream>>>(Xh, WTl, bl, WTr, br, xl, xr, N);
        gatv2_init   <<<initBlocks,     blk, 0, stream>>>(agg, m, denom, N);
        gatv2_logits <<<edgeWaveBlocks, blk, 0, stream>>>(xl, xr, att, ei, E, N, logit, m);
        gatv2_exp    <<<ehBlocks,       blk, 0, stream>>>(logit, m, ei, E, N, denom);
        gatv2_scatter<<<edgeWaveBlocks, blk, 0, stream>>>(xl, logit, denom, ei, E, N, agg);
        gatv2_final  <<<initBlocks,     blk, 0, stream>>>(agg, bias, outp, n128);
    }
}
